// Attention_4741643894804
// MI455X (gfx1250) — matrix operand order repaired and checked by reading, not compiled, not run on hardware
//
#include <hip/hip_runtime.h>
#include <hip/hip_bf16.h>

#define T_DIM 2048
#define B_DIM 32
#define N_DIM 1024
#define N2_DIM 512

typedef __bf16 v16bf __attribute__((ext_vector_type(16)));
typedef float  v8f   __attribute__((ext_vector_type(8)));
typedef unsigned int v4u __attribute__((ext_vector_type(4)));
typedef int v4i __attribute__((ext_vector_type(4)));
typedef int v8i __attribute__((ext_vector_type(8)));

// LDS staging (dynamic shared):
//  A buffer: 32 rows x (128 fp32 + 4 dword pad) = 32*132*4  = 16896 B   (132%64=4 -> conflict-free)
//  B buffer: 512 rows x (64 dwords + 4 dword pad) = 512*68*4 = 139264 B (68%64=4 -> conflict-free)
#define A_BYTES 16896
#define B_BYTES 139264
#define SMEM_BYTES (2 * A_BYTES + 2 * B_BYTES)   // 312320 B < 320 KB WGP LDS

// ---------------- Kernel 1: mean over time -> m[B,N] ----------------
__global__ __launch_bounds__(256) void mean_kernel(const float* __restrict__ hyp,
                                                   float* __restrict__ m) {
    int idx = blockIdx.x * 256 + threadIdx.x;     // [0, B*N)
    int b = idx / N_DIM, n = idx % N_DIM;
    float acc = 0.f;
    for (int t = 0; t < T_DIM; ++t) {
        acc += hyp[((size_t)t * B_DIM + b) * N_DIM + n];
    }
    m[idx] = acc * (1.0f / T_DIM);
}

// ---------------- Kernel 2: W_w fp32 -> bf16 ----------------
__global__ __launch_bounds__(256) void convert_w_kernel(const float* __restrict__ W,
                                                        __bf16* __restrict__ Wbf) {
    int i = blockIdx.x * 256 + threadIdx.x;       // [0, N2*N)
    Wbf[i] = (__bf16)W[i];
}

// ---------------- Kernel 3: v[b,k] = tanh(m.Wm_w^T + Wm_b) * Wh_w[k] ----------------
__global__ __launch_bounds__(256) void hm_v_kernel(const float* __restrict__ m,
                                                   const float* __restrict__ Wm_w,
                                                   const float* __restrict__ Wm_b,
                                                   const float* __restrict__ Wh_w,
                                                   float* __restrict__ v) {
    __shared__ float mlds[N_DIM];
    int b = blockIdx.x >> 1;
    int khalf = blockIdx.x & 1;
    for (int i = threadIdx.x; i < N_DIM; i += 256) mlds[i] = m[(size_t)b * N_DIM + i];
    __syncthreads();
    int k = khalf * 256 + threadIdx.x;
    const float* wrow = Wm_w + (size_t)k * N_DIM;
    float acc = 0.f;
#pragma unroll 4
    for (int n = 0; n < N_DIM; ++n) acc += mlds[n] * wrow[n];
    float hm = tanhf(acc + Wm_b[k]);
    v[(size_t)b * N2_DIM + k] = hm * Wh_w[k];
}

// ---------------- TDM helper: 2-D tile load global -> LDS via Tensor Data Mover ----
// Builds a D# (groups 0/1 per CDNA5 ISA 8.3/8.4) and issues TENSOR_LOAD_TO_LDS.
// pad_enable=1 so LDS rows are padded for bank-conflict-free fragment reads.
__device__ __forceinline__ void tdm_load_2d(unsigned lds_off, const void* gptr,
                                            unsigned ds_code,      // 0=1B,1=2B,2=4B
                                            unsigned tile_d0, unsigned tile_d1,
                                            unsigned long long stride0_elems,
                                            unsigned pad_int_code, unsigned pad_amt_code) {
    unsigned long long ga = (unsigned long long)(uintptr_t)gptr;
    v4u g0 = {0u, 0u, 0u, 0u};
    g0[0] = 1u;                                               // count=1, user descriptor
    g0[1] = lds_off;                                          // LDS byte address
    g0[2] = (unsigned)(ga & 0xffffffffu);                     // global_addr[31:0]
    g0[3] = (unsigned)((ga >> 32) & 0x01ffffffu) | 0x80000000u; // addr[56:32] | type=2
    const unsigned TD0 = 0x40000000u;                         // huge tensor dims: no OOB clip
    const unsigned TD1 = 0x00100000u;
    v8i g1 = {0, 0, 0, 0, 0, 0, 0, 0};
    g1[0] = (int)((ds_code << 16) | (1u << 20) |              // data_size | pad_enable
                  (pad_int_code << 22) | (pad_amt_code << 25));
    g1[1] = (int)(TD0 << 16);                                 // tensor_dim0[15:0]
    g1[2] = (int)((TD0 >> 16) | (TD1 << 16));                 // dim0[31:16] | dim1[15:0]
    g1[3] = (int)((TD1 >> 16) | (tile_d0 << 16));             // dim1[31:16] | tile_dim0
    g1[4] = (int)tile_d1;                                     // tile_dim1 (tile_dim2=0)
    g1[5] = (int)(unsigned)(stride0_elems & 0xffffffffu);     // tensor_dim0_stride lo
    g1[6] = (int)(unsigned)((stride0_elems >> 32) & 0xffffu); // stride hi | dim1_stride=0
    g1[7] = 0;
    v4i gz = {0, 0, 0, 0};
#if defined(__clang_major__) && (__clang_major__ >= 23)
    v8i gz8 = {0, 0, 0, 0, 0, 0, 0, 0};
    __builtin_amdgcn_tensor_load_to_lds(g0, g1, gz, gz, gz8, 0);
#else
    __builtin_amdgcn_tensor_load_to_lds(g0, g1, gz, gz, 0);
#endif
}

__device__ __forceinline__ void issue_tile_loads(const float* hyp, const __bf16* Wbf,
                                                 unsigned smem_off, int b, int t0blk,
                                                 int kcb) {
    const int buf = kcb & 1;
    // A: 32 t-rows x 128 fp32, row stride B*N elems; pad 4 dw per 128 dw -> 132 dw rows
    tdm_load_2d(smem_off + buf * A_BYTES,
                hyp + ((size_t)t0blk * B_DIM + b) * N_DIM + kcb * 128,
                2u, 128u, 32u, (unsigned long long)(B_DIM * N_DIM), 6u, 3u);
    // B: 512 n2-rows x 128 bf16, row stride N elems; pad 4 dw per 64 dw -> 68 dw rows
    tdm_load_2d(smem_off + 2 * A_BYTES + buf * B_BYTES,
                Wbf + (size_t)kcb * 128,
                1u, 128u, 512u, (unsigned long long)N_DIM, 5u, 3u);
}

// ---------------- Kernel 4: fused GEMM + tanh + scale + k-reduce -> score[B,T] ----------------
// grid: (T/32, B); block: 256 threads = 8 waves.
// wave w: tt = w>>2 (t-tile of 16 rows), kq = w&3 (128 k-columns = 8 WMMA tiles)
// TDM double-buffers A (fp32) and B (bf16) tiles in LDS; waves consume via ds_load + WMMA.
__global__ __launch_bounds__(256) void score_gemm_kernel(const float* __restrict__ hyp,
                                                         const __bf16* __restrict__ Wbf,
                                                         const float* __restrict__ W_b,
                                                         const float* __restrict__ v,
                                                         const float* __restrict__ Wh_b,
                                                         float* __restrict__ score) {
    extern __shared__ char smem[];
    const int b     = blockIdx.y;
    const int t0blk = blockIdx.x * 32;
    const int wave  = threadIdx.x >> 5;
    const int lane  = threadIdx.x & 31;
    const int tt    = wave >> 2;
    const int kq    = wave & 3;
    const int hi16  = lane >> 4;   // 0: lanes 0-15, 1: lanes 16-31
    const int l16   = lane & 15;
    const unsigned smem_off = (unsigned)(uintptr_t)(void*)smem;  // LDS byte offset

    if (wave == 0) {
        issue_tile_loads(hyp, Wbf, smem_off, b, t0blk, 0);
        issue_tile_loads(hyp, Wbf, smem_off, b, t0blk, 1);
    }

    v8f C[8] = {};

    for (int kcb = 0; kcb < 8; ++kcb) {
        if (wave == 0) {
            if (kcb < 7) __builtin_amdgcn_s_wait_tensorcnt((short)2);  // buffer kcb ready
            else         __builtin_amdgcn_s_wait_tensorcnt((short)0);
        }
        __syncthreads();

        const float*  aBuf = (const float*)(smem + (kcb & 1) * A_BYTES);
        const __bf16* bBuf = (const __bf16*)(smem + 2 * A_BYTES + (kcb & 1) * B_BYTES);
        const float*  arow = aBuf + (tt * 16 + l16) * 132;   // padded row stride

#pragma unroll
        for (int kc = 0; kc < 128; kc += 32) {
            // ---- A fragment from LDS (fp32 -> bf16 in-register), 16x32 bf16 A layout ----
            // lanes 0-15: K = kc+0..7 (elems 0-7), kc+16..23 (elems 8-15)
            // lanes 16-31: K = kc+8..15,           kc+24..31
            const float* ap = arow + kc + hi16 * 8;
            float4 f0 = ((const float4*)ap)[0];
            float4 f1 = ((const float4*)ap)[1];
            float4 f2 = ((const float4*)(ap + 16))[0];
            float4 f3 = ((const float4*)(ap + 16))[1];
            float af[16] = {f0.x, f0.y, f0.z, f0.w, f1.x, f1.y, f1.z, f1.w,
                            f2.x, f2.y, f2.z, f2.w, f3.x, f3.y, f3.z, f3.w};
            v16bf A;
#pragma unroll
            for (int i = 0; i < 16; ++i) A[i] = (__bf16)af[i];

#pragma unroll
            for (int kt = 0; kt < 8; ++kt) {
                // ---- B fragment from LDS: column n2 = padded row n2 (136 bf16 stride) ----
                // lanes 0-15: K = kc+0..7, kc+16..23 ; lanes 16-31: K = kc+8..15, kc+24..31
                const int n2 = kq * 128 + kt * 16 + l16;
                union { uint4 u[2]; v16bf vv; } Bf;
                const uint4* bp = (const uint4*)(bBuf + (size_t)n2 * 136 + kc + hi16 * 8);
                Bf.u[0] = bp[0];
                Bf.u[1] = bp[2];
                C[kt] = __builtin_amdgcn_wmma_f32_16x16x32_bf16(
                    false, A, false, Bf.vv, (short)0, C[kt], false, false);
            }
        }
        __syncthreads();   // all waves done with buffer kcb&1 before TDM overwrites it
        if (wave == 0 && kcb < 6) issue_tile_loads(hyp, Wbf, smem_off, b, t0blk, kcb + 2);
    }

    // ---- epilogue: tanh(+bias)*v, reduce over k within the wave ----
    float rowacc[8];
#pragma unroll
    for (int r = 0; r < 8; ++r) rowacc[r] = 0.f;
#pragma unroll
    for (int kt = 0; kt < 8; ++kt) {
        const int kg = kq * 128 + kt * 16 + l16;  // lane's column of C tile
        const float bias = W_b[kg];
        const float vs   = v[(size_t)b * N2_DIM + kg];
#pragma unroll
        for (int r = 0; r < 8; ++r) {
            rowacc[r] += tanhf(C[kt][r] + bias) * vs;   // row M = r + hi16*8
        }
    }
    // reduce across the 16 lanes of each half-group (C layout: col = lane%16)
#pragma unroll
    for (int r = 0; r < 8; ++r) {
        float s = rowacc[r];
        s += __shfl_xor(s, 1, 16);
        s += __shfl_xor(s, 2, 16);
        s += __shfl_xor(s, 4, 16);
        s += __shfl_xor(s, 8, 16);
        rowacc[r] = s;
    }

    __shared__ float sscore[32];
    if (threadIdx.x < 32) sscore[threadIdx.x] = 0.f;
    __syncthreads();
    if (l16 == 0) {
#pragma unroll
        for (int r = 0; r < 8; ++r) {
            atomicAdd(&sscore[tt * 16 + hi16 * 8 + r], rowacc[r]);  // ds_add_f32
        }
    }
    __syncthreads();
    if (threadIdx.x < 32) {
        score[(size_t)b * T_DIM + t0blk + threadIdx.x] = sscore[threadIdx.x] + Wh_b[0];
    }
}

// ---------------- Kernel 5: softmax over T per batch ----------------
__global__ __launch_bounds__(256) void softmax_kernel(const float* __restrict__ score,
                                                      float* __restrict__ a) {
    int b = blockIdx.x;
    __shared__ float red[256];
    const float* s = score + (size_t)b * T_DIM;
    float mx = -1e30f;
    for (int t = threadIdx.x; t < T_DIM; t += 256) mx = fmaxf(mx, s[t]);
    red[threadIdx.x] = mx;
    __syncthreads();
    for (int off = 128; off > 0; off >>= 1) {
        if (threadIdx.x < off) red[threadIdx.x] = fmaxf(red[threadIdx.x], red[threadIdx.x + off]);
        __syncthreads();
    }
    mx = red[0];
    __syncthreads();
    float sum = 0.f;
    for (int t = threadIdx.x; t < T_DIM; t += 256) {
        float e = __expf(s[t] - mx);
        a[(size_t)b * T_DIM + t] = e;
        sum += e;
    }
    red[threadIdx.x] = sum;
    __syncthreads();
    for (int off = 128; off > 0; off >>= 1) {
        if (threadIdx.x < off) red[threadIdx.x] += red[threadIdx.x + off];
        __syncthreads();
    }
    float inv = 1.0f / red[0];
    for (int t = threadIdx.x; t < T_DIM; t += 256) a[(size_t)b * T_DIM + t] *= inv;
}

// ---------------- Kernel 6: c[b,n] = sum_t a[b,t] * x[b,t,n] ----------------
__global__ __launch_bounds__(256) void wsum_kernel(const float* __restrict__ hyp,
                                                   const float* __restrict__ a,
                                                   float* __restrict__ out) {
    int idx = blockIdx.x * 256 + threadIdx.x;     // [0, B*N)
    int b = idx / N_DIM, n = idx % N_DIM;
    const float* ab = a + (size_t)b * T_DIM;
    float acc = 0.f;
    for (int t = 0; t < T_DIM; ++t) {
        acc += ab[t] * hyp[((size_t)t * B_DIM + b) * N_DIM + n];
    }
    out[idx] = acc;
}

extern "C" void kernel_launch(void* const* d_in, const int* in_sizes, int n_in,
                              void* d_out, int out_size, void* d_ws, size_t ws_size,
                              hipStream_t stream) {
    const float* hyp  = (const float*)d_in[0];
    const float* W_w  = (const float*)d_in[1];
    const float* W_b  = (const float*)d_in[2];
    const float* Wm_w = (const float*)d_in[3];
    const float* Wm_b = (const float*)d_in[4];
    const float* Wh_w = (const float*)d_in[5];
    const float* Wh_b = (const float*)d_in[6];
    float* out = (float*)d_out;

    // workspace layout (all offsets 256B-aligned); total ~1.7 MB
    char* ws = (char*)d_ws;
    float*  m     = (float*)(ws);                                    // B*N*4   = 131072
    float*  vv    = (float*)(ws + 131072);                           // B*N2*4  = 65536
    float*  score = (float*)(ws + 131072 + 65536);                   // B*T*4   = 262144
    float*  attn  = (float*)(ws + 131072 + 65536 + 262144);          // B*T*4   = 262144
    __bf16* Wbf   = (__bf16*)(ws + 131072 + 65536 + 262144 + 262144);// N2*N*2  = 1048576

    mean_kernel<<<(B_DIM * N_DIM) / 256, 256, 0, stream>>>(hyp, m);
    convert_w_kernel<<<(N2_DIM * N_DIM) / 256, 256, 0, stream>>>(W_w, Wbf);
    hm_v_kernel<<<B_DIM * (N2_DIM / 256), 256, 0, stream>>>(m, Wm_w, Wm_b, Wh_w, vv);
    score_gemm_kernel<<<dim3(T_DIM / 32, B_DIM), 256, SMEM_BYTES, stream>>>(
        hyp, Wbf, W_b, vv, Wh_b, score);
    softmax_kernel<<<B_DIM, 256, 0, stream>>>(score, attn);
    wsum_kernel<<<(B_DIM * N_DIM) / 256, 256, 0, stream>>>(hyp, attn, out);
}
